// LongformerEncoder_32255204393617
// MI455X (gfx1250) — matrix operand order repaired and checked by reading, not compiled, not run on hardware
//
#include <hip/hip_runtime.h>

// ---------------- problem constants ----------------
constexpr int BB  = 2;      // batch
constexpr int SS  = 4096;   // sequence
constexpr int DD  = 768;    // model dim
constexpr int NH  = 12;     // heads
constexpr int DHD = 64;     // head dim
constexpr int NL  = 12;     // layers
constexpr int WIN = 256;    // band half-window
constexpr int FFD = 3072;   // FF dim
constexpr int HID = 384;    // LSTM hidden

// ---------------- vector types ----------------
typedef __attribute__((ext_vector_type(16))) __bf16 v16bf;
typedef __attribute__((ext_vector_type(8)))  __bf16 v8bf;
typedef __attribute__((ext_vector_type(8)))  float  v8f;
typedef __attribute__((ext_vector_type(4)))  unsigned int u32x4;
typedef __attribute__((ext_vector_type(8)))  int    i32x8;
typedef __attribute__((ext_vector_type(4)))  int    i32x4;

union BFU16 { v16bf v; v8bf h[2]; };

__device__ __forceinline__ __bf16 f2bf(float f) {
  union { float f; unsigned u; } c; c.f = f;
  unsigned r = (c.u + 0x7FFFu + ((c.u >> 16) & 1u)) >> 16;
  union { unsigned short s; __bf16 b; } o; o.s = (unsigned short)r;
  return o.b;
}

__device__ __forceinline__ v8f vzero8() {
  v8f z;
  #pragma unroll
  for (int i = 0; i < 8; ++i) z[i] = 0.f;
  return z;
}

__device__ __forceinline__ v8f wmma_bf16(v16bf a, v16bf b, v8f c) {
  return __builtin_amdgcn_wmma_f32_16x16x32_bf16(false, a, false, b, (short)0, c, false, false);
}

// A fragment: 16x32 bf16 tile; lane m = lane&15, 16 contiguous K by lane>>4
__device__ __forceinline__ v16bf frag_a(const __bf16* tile, int row0, int stride, int kk, int lane) {
  const int r  = row0 + (lane & 15);
  const int ko = (lane >> 4) * 16;
  BFU16 u;
  u.h[0] = *(const v8bf*)(tile + r * stride + kk + ko);
  u.h[1] = *(const v8bf*)(tile + r * stride + kk + ko + 8);
  return u.v;
}

// B fragment: 32x16 bf16; lane n = lane&15, 16 contiguous K (tileT is [N][K])
__device__ __forceinline__ v16bf frag_b(const __bf16* tileT, int col0, int stride, int kk, int lane) {
  const int n  = col0 + (lane & 15);
  const int ko = (lane >> 4) * 16;
  return *(const v16bf*)(tileT + n * stride + kk + ko);
}

// TDM: async-DMA a [rows x 32] bf16 tile (row stride = ld elements) into LDS
__device__ __forceinline__ void tdm_load_tile(const __bf16* gsrc, unsigned ldsOff,
                                              int ld, int totRows, int tileRows) {
  unsigned long long ga = (unsigned long long)gsrc;
  u32x4 g0; i32x8 g1; i32x4 gz; i32x8 gz8;
  g0[0] = 1u;                                             // count=1, no gather
  g0[1] = ldsOff;                                         // lds_addr
  g0[2] = (unsigned)(ga & 0xFFFFFFFFull);                 // global_addr lo
  g0[3] = (unsigned)((ga >> 32) & 0x1FFFFFFull) | (2u << 30); // hi + type=2
  g1[0] = (int)(1u << 16);                                // data_size=2B
  g1[1] = (int)((ld & 0xFFFF) << 16);                     // tensor_dim0 lo16
  g1[2] = (int)(((ld >> 16) & 0xFFFF) | ((totRows & 0xFFFF) << 16));
  g1[3] = (int)(((totRows >> 16) & 0xFFFF) | (32u << 16)); // tile_dim0=32
  g1[4] = tileRows;                                        // tile_dim1, tile_dim2=0
  g1[5] = ld;                                              // tensor_dim0_stride lo32
  g1[6] = 0; g1[7] = 0;
  gz[0] = 0; gz[1] = 0; gz[2] = 0; gz[3] = 0;
  #pragma unroll
  for (int i = 0; i < 8; ++i) gz8[i] = 0;
  __builtin_amdgcn_tensor_load_to_lds(g0, g1, gz, gz, gz8, 0);
}

// =====================================================================
// GEMM: out = act(A(bf16)[M,K] @ W(f32) + bias (+resid)) ; W is [K][N]
// flags: bit0 = exact GELU, bit1 = W stored transposed as [N][K]
// =====================================================================
__global__ __launch_bounds__(256) void gemm_bf16_kernel(
    const __bf16* __restrict__ A, const float* __restrict__ Wg,
    const float* __restrict__ bias, const float* __restrict__ resid,
    float* __restrict__ outF, __bf16* __restrict__ outB,
    int M, int N, int K, int flags)
{
  constexpr int BM = 128, BN = 128, BK = 32;
  __shared__ __align__(64) __bf16 As[BM * BK];
  __shared__ __align__(64) __bf16 Bt[BN * BK];

  const int tid  = threadIdx.x;
  const int lane = tid & 31;
  const int wave = tid >> 5;
  const int wm   = wave >> 1;   // 0..3  -> 32 rows
  const int wn   = wave & 1;    // 0..1  -> 64 cols
  const int m0   = blockIdx.y * BM;
  const int n0   = blockIdx.x * BN;

  v8f acc[2][4];
  #pragma unroll
  for (int mi = 0; mi < 2; ++mi)
    #pragma unroll
    for (int ni = 0; ni < 4; ++ni) acc[mi][ni] = vzero8();

  for (int k0 = 0; k0 < K; k0 += BK) {
    // --- A tile via Tensor Data Mover (wave 0 issues; others stage W) ---
    if (wave == 0) {
      tdm_load_tile(A + (size_t)m0 * K + k0,
                    (unsigned)(unsigned long long)(const void*)As, K, M, BM);
    }
    // --- W tile -> bf16, transposed into Bt[N][K] ---
    if (flags & 2) {                     // W is [N][K]
      const int n = tid >> 1, half = (tid & 1) * 16;
      const float* src = Wg + (size_t)(n0 + n) * K + k0 + half;
      __bf16* dst = Bt + n * BK + half;
      #pragma unroll
      for (int j = 0; j < 16; ++j) dst[j] = f2bf(src[j]);
    } else {                             // W is [K][N]
      const int k = tid & 31, ng = tid >> 5;
      const float* src = Wg + (size_t)(k0 + k) * N + n0 + ng * 16;
      #pragma unroll
      for (int j = 0; j < 16; ++j) Bt[(ng * 16 + j) * BK + k] = f2bf(src[j]);
    }
    // prefetch next weight tile into GL2
    __builtin_prefetch(Wg + ((flags & 2) ? ((size_t)n0 * K + k0 + BK)
                                         : ((size_t)(k0 + BK) * N + n0)), 0, 1);
    if (wave == 0) __builtin_amdgcn_s_wait_tensorcnt(0);
    __syncthreads();

    v16bf af[2], bfv[4];
    #pragma unroll
    for (int mi = 0; mi < 2; ++mi) af[mi] = frag_a(As, wm * 32 + mi * 16, BK, 0, lane);
    #pragma unroll
    for (int ni = 0; ni < 4; ++ni) bfv[ni] = frag_b(Bt, wn * 64 + ni * 16, BK, 0, lane);
    #pragma unroll
    for (int mi = 0; mi < 2; ++mi)
      #pragma unroll
      for (int ni = 0; ni < 4; ++ni)
        acc[mi][ni] = wmma_bf16(af[mi], bfv[ni], acc[mi][ni]);
    __syncthreads();
  }

  const bool gelu = flags & 1;
  #pragma unroll
  for (int mi = 0; mi < 2; ++mi)
    #pragma unroll
    for (int ni = 0; ni < 4; ++ni) {
      const int nn = n0 + wn * 64 + ni * 16 + (lane & 15);
      const float bvv = bias ? bias[nn] : 0.f;
      #pragma unroll
      for (int r = 0; r < 8; ++r) {
        const int mm = m0 + wm * 32 + mi * 16 + (lane >> 4) * 8 + r;
        float v = acc[mi][ni][r] + bvv;
        if (resid) v += resid[(size_t)mm * N + nn];
        if (gelu)  v = 0.5f * v * (1.f + erff(v * 0.70710678f));
        if (outF) outF[(size_t)mm * N + nn] = v;
        if (outB) outB[(size_t)mm * N + nn] = f2bf(v);
      }
    }
}

// =====================================================================
// Sliding-window attention (flash-style), per (64-query tile, head, b)
// =====================================================================
__global__ __launch_bounds__(128) void band_attn_kernel(
    const __bf16* __restrict__ qB, const __bf16* __restrict__ kB,
    const __bf16* __restrict__ vB, const int* __restrict__ tok,
    __bf16* __restrict__ ctxB)
{
  __shared__ __align__(64) __bf16 Qs[64 * 64];
  __shared__ __align__(64) __bf16 Ks[64 * 64];
  __shared__ __align__(64) __bf16 Vt[64 * 64];   // transposed [dh][key]
  __shared__ __align__(64) float  Sc[64 * 64];
  __shared__ __align__(64) __bf16 Pb[64 * 64];
  __shared__ float Mrow[64], Lrow[64], Rsc[64];

  const int q0 = blockIdx.x * 64, hh = blockIdx.y, bb = blockIdx.z;
  const int tid = threadIdx.x, lane = tid & 31, wave = tid >> 5;

  { // stage Q (64 rows x 64 bf16)
    const int r = tid >> 1, c = (tid & 1) * 32;
    const v8bf* s = (const v8bf*)(qB + ((size_t)(bb * SS + q0 + r) * DD) + hh * 64 + c);
    v8bf* d = (v8bf*)(Qs + r * 64 + c);
    #pragma unroll
    for (int j = 0; j < 4; ++j) d[j] = s[j];
  }
  if (tid < 64) { Mrow[tid] = -1e30f; Lrow[tid] = 0.f; }

  v8f O[4];
  #pragma unroll
  for (int ni = 0; ni < 4; ++ni) O[ni] = vzero8();

  for (int slab = 0; slab < 10; ++slab) {
    const int ks0 = q0 - 256 + slab * 64;
    __syncthreads();
    { // stage K slab and transposed V slab (zero out-of-range)
      const int r = tid >> 1, c = (tid & 1) * 32;
      const int key = ks0 + r;
      if (key >= 0 && key < SS) {
        const v8bf* sk = (const v8bf*)(kB + ((size_t)(bb * SS + key) * DD) + hh * 64 + c);
        v8bf* dk = (v8bf*)(Ks + r * 64 + c);
        #pragma unroll
        for (int j = 0; j < 4; ++j) dk[j] = sk[j];
        const __bf16* sv = vB + ((size_t)(bb * SS + key) * DD) + hh * 64 + c;
        #pragma unroll
        for (int j = 0; j < 32; ++j) Vt[(c + j) * 64 + r] = sv[j];
      } else {
        const __bf16 z = f2bf(0.f);
        #pragma unroll
        for (int j = 0; j < 32; ++j) { Ks[r * 64 + c + j] = z; Vt[(c + j) * 64 + r] = z; }
      }
    }
    __syncthreads();

    // scores: S[64x64] = Q @ K^T  (wave owns 16 rows, all 64 cols)
    #pragma unroll
    for (int ni = 0; ni < 4; ++ni) {
      v8f s = vzero8();
      #pragma unroll
      for (int kk = 0; kk < 64; kk += 32) {
        v16bf a  = frag_a(Qs, wave * 16, 64, kk, lane);
        v16bf bv = frag_b(Ks, ni * 16, 64, kk, lane);
        s = wmma_bf16(a, bv, s);
      }
      const int nn  = ni * 16 + (lane & 15);
      const int key = ks0 + nn;
      bool kval = false;
      if (key >= 0 && key < SS) kval = (tok[(size_t)bb * SS + key] != 0);
      #pragma unroll
      for (int r = 0; r < 8; ++r) {
        const int mm = wave * 16 + (lane >> 4) * 8 + r;
        const int q  = q0 + mm;
        const bool valid = kval && (key >= q - WIN) && (key <= q + WIN);
        Sc[mm * 64 + nn] = valid ? s[r] * 0.125f : -1e9f;
      }
    }
    __syncthreads();

    // online softmax update per row
    if (tid < 64) {
      float mOld = Mrow[tid], mNew = mOld;
      for (int c = 0; c < 64; ++c) mNew = fmaxf(mNew, Sc[tid * 64 + c]);
      const float scl = __expf(mOld - mNew);
      float lsum = 0.f;
      for (int c = 0; c < 64; ++c) {
        const float pe = __expf(Sc[tid * 64 + c] - mNew);
        lsum += pe;
        Pb[tid * 64 + c] = f2bf(pe);
      }
      Lrow[tid] = Lrow[tid] * scl + lsum;
      Mrow[tid] = mNew;
      Rsc[tid]  = scl;
    }
    __syncthreads();

    // O = O*scale + P @ V
    #pragma unroll
    for (int ni = 0; ni < 4; ++ni) {
      #pragma unroll
      for (int r = 0; r < 8; ++r) {
        const int mm = wave * 16 + (lane >> 4) * 8 + r;
        O[ni][r] *= Rsc[mm];
      }
      #pragma unroll
      for (int kk = 0; kk < 64; kk += 32) {
        v16bf a  = frag_a(Pb, wave * 16, 64, kk, lane);
        v16bf bv = frag_b(Vt, ni * 16, 64, kk, lane);
        O[ni] = wmma_bf16(a, bv, O[ni]);
      }
    }
  }
  __syncthreads();

  #pragma unroll
  for (int ni = 0; ni < 4; ++ni) {
    const int dh = ni * 16 + (lane & 15);
    #pragma unroll
    for (int r = 0; r < 8; ++r) {
      const int mm = wave * 16 + (lane >> 4) * 8 + r;
      const float l = fmaxf(Lrow[mm], 1e-20f);
      ctxB[((size_t)(bb * SS + q0 + mm) * DD) + hh * 64 + dh] = f2bf(O[ni][r] / l);
    }
  }
}

// =====================================================================
// embedding gather + LayerNorm ;  residual + LayerNorm
// =====================================================================
__global__ __launch_bounds__(256) void embed_ln_kernel(
    const int* __restrict__ x, const float* __restrict__ embW,
    const float* __restrict__ embP, const float* __restrict__ g,
    const float* __restrict__ bta, float* __restrict__ outF, __bf16* __restrict__ outB)
{
  __shared__ float red[256];
  const int row = blockIdx.x, tid = threadIdx.x;
  const int s = row % SS;
  const int tokid = x[row];
  const size_t base = (size_t)row * DD;
  float v[3];
  #pragma unroll
  for (int j = 0; j < 3; ++j) {
    const int i = tid + j * 256;
    v[j] = embW[(size_t)tokid * DD + i] + embP[(size_t)s * DD + i];
  }
  float sm = v[0] + v[1] + v[2];
  red[tid] = sm; __syncthreads();
  for (int o = 128; o > 0; o >>= 1) { if (tid < o) red[tid] += red[tid + o]; __syncthreads(); }
  const float mean = red[0] * (1.f / 768.f);
  __syncthreads();
  float q = 0.f;
  #pragma unroll
  for (int j = 0; j < 3; ++j) { const float d = v[j] - mean; q += d * d; }
  red[tid] = q; __syncthreads();
  for (int o = 128; o > 0; o >>= 1) { if (tid < o) red[tid] += red[tid + o]; __syncthreads(); }
  const float rstd = rsqrtf(red[0] * (1.f / 768.f) + 1e-5f);
  #pragma unroll
  for (int j = 0; j < 3; ++j) {
    const int i = tid + j * 256;
    const float o = (v[j] - mean) * rstd * g[i] + bta[i];
    outF[base + i] = o; outB[base + i] = f2bf(o);
  }
}

__global__ __launch_bounds__(256) void resid_ln_kernel(
    const float* __restrict__ res, const float* __restrict__ xin,
    const float* __restrict__ g, const float* __restrict__ bta,
    float* __restrict__ outF, __bf16* __restrict__ outB)
{
  __shared__ float red[256];
  const int row = blockIdx.x, tid = threadIdx.x;
  const size_t base = (size_t)row * DD;
  float v[3];
  #pragma unroll
  for (int j = 0; j < 3; ++j) { const int i = tid + j * 256; v[j] = res[base + i] + xin[base + i]; }
  float sm = v[0] + v[1] + v[2];
  red[tid] = sm; __syncthreads();
  for (int o = 128; o > 0; o >>= 1) { if (tid < o) red[tid] += red[tid + o]; __syncthreads(); }
  const float mean = red[0] * (1.f / 768.f);
  __syncthreads();
  float q = 0.f;
  #pragma unroll
  for (int j = 0; j < 3; ++j) { const float d = v[j] - mean; q += d * d; }
  red[tid] = q; __syncthreads();
  for (int o = 128; o > 0; o >>= 1) { if (tid < o) red[tid] += red[tid + o]; __syncthreads(); }
  const float rstd = rsqrtf(red[0] * (1.f / 768.f) + 1e-5f);
  #pragma unroll
  for (int j = 0; j < 3; ++j) {
    const int i = tid + j * 256;
    const float o = (v[j] - mean) * rstd * g[i] + bta[i];
    outF[base + i] = o; outB[base + i] = f2bf(o);
  }
}

// =====================================================================
// persistent bidirectional LSTM (blockIdx.x = direction)
// =====================================================================
__device__ __forceinline__ float sigf(float x) { return 1.f / (1.f + __expf(-x)); }

__global__ __launch_bounds__(256) void lstm_kernel(
    const float* __restrict__ xgF, const float* __restrict__ xgB,
    const float* __restrict__ whF, const float* __restrict__ whB,
    float* __restrict__ out)
{
  __shared__ float hbuf[2 * HID], cbuf[2 * HID], gl[2 * 4 * HID];
  const int dir = blockIdx.x, tid = threadIdx.x;
  const float* xg = dir ? xgB : xgF;
  const float* wh = dir ? whB : whF;
  for (int i = tid; i < 2 * HID; i += 256) { hbuf[i] = 0.f; cbuf[i] = 0.f; }
  __syncthreads();
  for (int step = 0; step < SS; ++step) {
    const int t = dir ? (SS - 1 - step) : step;
    #pragma unroll
    for (int j = 0; j < 12; ++j) {            // 2*1536 outputs / 256 threads
      const int idx = tid + j * 256;
      const int b2 = idx / (4 * HID), n = idx - b2 * 4 * HID;
      const float* wr = wh + (size_t)n * HID;
      const float* hr = hbuf + b2 * HID;
      float a = xg[((size_t)b2 * SS + t) * (4 * HID) + n];
      for (int k2 = 0; k2 < HID; ++k2) a += hr[k2] * wr[k2];
      gl[idx] = a;
    }
    __syncthreads();
    if (tid < 2 * HID) {
      const int b2 = tid / HID, u = tid - b2 * HID;
      const float* g = gl + b2 * 4 * HID;
      const float gi = g[u], gf = g[HID + u], gg = g[2 * HID + u], go = g[3 * HID + u];
      const float c = sigf(gf) * cbuf[tid] + sigf(gi) * tanhf(gg);
      const float hv = sigf(go) * tanhf(c);
      cbuf[tid] = c; hbuf[tid] = hv;
      if ((t & 63) == 0)
        out[((size_t)(b2 * 64 + (t >> 6))) * 768 + dir * HID + u] = hv;
    }
    __syncthreads();
  }
}

// =====================================================================
// driver
// =====================================================================
extern "C" void kernel_launch(void* const* d_in, const int* in_sizes, int n_in,
                              void* d_out, int out_size, void* d_ws, size_t ws_size,
                              hipStream_t stream)
{
  const int*   x     = (const int*)  d_in[0];
  const float* emb_w = (const float*)d_in[1];
  const float* emb_p = (const float*)d_in[2];
  const float* ln_eg = (const float*)d_in[3];
  const float* ln_eb = (const float*)d_in[4];
  const float* Wq    = (const float*)d_in[5];
  const float* bq    = (const float*)d_in[6];
  const float* Wk    = (const float*)d_in[7];
  const float* bk    = (const float*)d_in[8];
  const float* Wv    = (const float*)d_in[9];
  const float* bv    = (const float*)d_in[10];
  const float* Wo    = (const float*)d_in[11];
  const float* bo    = (const float*)d_in[12];
  const float* ln1g  = (const float*)d_in[13];
  const float* ln1b  = (const float*)d_in[14];
  const float* W1    = (const float*)d_in[15];
  const float* bf1   = (const float*)d_in[16];
  const float* W2    = (const float*)d_in[17];
  const float* bf2   = (const float*)d_in[18];
  const float* ln2g  = (const float*)d_in[19];
  const float* ln2b  = (const float*)d_in[20];
  const float* wi_f  = (const float*)d_in[21];
  const float* wh_f  = (const float*)d_in[22];
  const float* b_f   = (const float*)d_in[23];
  const float* wi_b  = (const float*)d_in[24];
  const float* wh_b  = (const float*)d_in[25];
  const float* b_b   = (const float*)d_in[26];
  float* out = (float*)d_out;

  const size_t Mrows = (size_t)BB * SS;   // 8192
  char* p = (char*)d_ws;
  auto take = [&](size_t bytes) { char* r = p; p += (bytes + 255) & ~(size_t)255; return (void*)r; };
  float*  hF   = (float*) take(Mrows * DD * 4);
  __bf16* hB   = (__bf16*)take(Mrows * DD * 2);
  __bf16* qBf  = (__bf16*)take(Mrows * DD * 2);
  __bf16* kBf  = (__bf16*)take(Mrows * DD * 2);
  __bf16* vBf  = (__bf16*)take(Mrows * DD * 2);
  __bf16* ctxB = (__bf16*)take(Mrows * DD * 2);
  float*  tF   = (float*) take(Mrows * DD * 4);
  __bf16* ffB  = (__bf16*)take(Mrows * FFD * 2);
  float*  xgF  = (float*) take(Mrows * 4 * HID * 4);
  float*  xgB  = (float*) take(Mrows * 4 * HID * 4);

  const dim3 gProj(DD / 128, (unsigned)(Mrows / 128));    // (6, 64)
  const dim3 gFF1(FFD / 128, (unsigned)(Mrows / 128));    // (24, 64)
  const dim3 gGate((4 * HID) / 128, (unsigned)(Mrows / 128)); // (12, 64)
  const dim3 gAttn(SS / 64, NH, BB);                      // (64, 12, 2)

  embed_ln_kernel<<<(unsigned)Mrows, 256, 0, stream>>>(x, emb_w, emb_p, ln_eg, ln_eb, hF, hB);

  for (int l = 0; l < NL; ++l) {
    const size_t od = (size_t)l * DD * DD;
    gemm_bf16_kernel<<<gProj, 256, 0, stream>>>(hB, Wq + od, bq + (size_t)l * DD, nullptr,
                                                nullptr, qBf, (int)Mrows, DD, DD, 0);
    gemm_bf16_kernel<<<gProj, 256, 0, stream>>>(hB, Wk + od, bk + (size_t)l * DD, nullptr,
                                                nullptr, kBf, (int)Mrows, DD, DD, 0);
    gemm_bf16_kernel<<<gProj, 256, 0, stream>>>(hB, Wv + od, bv + (size_t)l * DD, nullptr,
                                                nullptr, vBf, (int)Mrows, DD, DD, 0);
    band_attn_kernel<<<gAttn, 128, 0, stream>>>(qBf, kBf, vBf, x, ctxB);
    gemm_bf16_kernel<<<gProj, 256, 0, stream>>>(ctxB, Wo + od, bo + (size_t)l * DD, nullptr,
                                                tF, nullptr, (int)Mrows, DD, DD, 0);
    resid_ln_kernel<<<(unsigned)Mrows, 256, 0, stream>>>(hF, tF, ln1g + (size_t)l * DD,
                                                         ln1b + (size_t)l * DD, hF, hB);
    gemm_bf16_kernel<<<gFF1, 256, 0, stream>>>(hB, W1 + (size_t)l * DD * FFD,
                                               bf1 + (size_t)l * FFD, nullptr,
                                               nullptr, ffB, (int)Mrows, FFD, DD, 1 /*gelu*/);
    gemm_bf16_kernel<<<gProj, 256, 0, stream>>>(ffB, W2 + (size_t)l * FFD * DD,
                                                bf2 + (size_t)l * DD, nullptr,
                                                tF, nullptr, (int)Mrows, DD, FFD, 0);
    resid_ln_kernel<<<(unsigned)Mrows, 256, 0, stream>>>(hF, tF, ln2g + (size_t)l * DD,
                                                         ln2b + (size_t)l * DD, hF, hB);
  }

  // LSTM input-gate GEMMs (weights stored [4H][D] -> transposed flag)
  gemm_bf16_kernel<<<gGate, 256, 0, stream>>>(hB, wi_f, b_f, nullptr, xgF, nullptr,
                                              (int)Mrows, 4 * HID, DD, 2);
  gemm_bf16_kernel<<<gGate, 256, 0, stream>>>(hB, wi_b, b_b, nullptr, xgB, nullptr,
                                              (int)Mrows, 4 * HID, DD, 2);
  lstm_kernel<<<2, 256, 0, stream>>>(xgF, xgB, wh_f, wh_b, out);
}
